// UniCrystalFormer_18975165514256
// MI455X (gfx1250) — compile-verified, matrix operand order checked by reading, not run on hardware
//
#include <hip/hip_runtime.h>

#define N_NODES 20000
#define N_EDGES 160000
#define N_GRAPH 128
#define AIF 92
#define AIFP 96   // AIF padded to multiple of 32
#define C 128
#define H 4
#define HC 512
#define C3 384
#define NLAYERS 5

#define ACT_NONE 0
#define ACT_SOFTPLUS 1
#define ACT_SILU 2

typedef __attribute__((ext_vector_type(16))) _Float16 v16h;
typedef __attribute__((ext_vector_type(8)))  _Float16 v8h;
typedef __attribute__((ext_vector_type(8)))  float    v8f;

__device__ __forceinline__ v8f wmma16(v16h a, v16h b, v8f c) {
  return __builtin_amdgcn_wmma_f32_16x16x32_f16(false, a, false, b, (short)0, c,
                                                false, false);
}

// One 16x32 f16 fragment from a row-major matrix whose rows are contiguous in
// K (works for A, and for B when the weight is stored transposed N x K).
// ISA 7.12.2 layout: lanes 0-15 hold row (lane&15); elems 0..7 -> K=k0..k0+7,
// elems 8..15 -> K=k0+16..k0+23; lanes 16-31 shift K by +8.
// Two 16-byte vector loads per lane.
__device__ __forceinline__ v16h frag_row(const _Float16* base, int ld, int row,
                                         int k0, int lane) {
  const int half = lane >> 4;
  const _Float16* p = base + (size_t)row * ld + k0 + half * 8;
  v8h lo = *(const v8h*)(p);
  v8h hi = *(const v8h*)(p + 16);
  v16h f;
#pragma unroll
  for (int e = 0; e < 8; ++e) {
    f[e] = lo[e];
    f[e + 8] = hi[e];
  }
  return f;
}

// ---- generic WMMA GEMM -----------------------------------------------------
// Out(MxN,f16) = act(A(MxK,f16) @ Wt(NxK,f16)^T + bias). One wave per block,
// 16x64 output strip (A fragment reused across 4 WMMAs per k-step).
// M % 16 == 0, N % 64 == 0, K % 32 == 0.
__global__ void gemm16(const _Float16* __restrict__ A,
                       const _Float16* __restrict__ Wt,
                       const float* __restrict__ bias,
                       _Float16* __restrict__ Out, int M, int K, int Ncols,
                       int act) {
  const int lane = threadIdx.x;
  const int m0 = blockIdx.x * 16;
  const int n0 = blockIdx.y * 64;
  const int mrow = m0 + (lane & 15);
  const int nrow = n0 + (lane & 15);
  v8f acc0 = {}, acc1 = {}, acc2 = {}, acc3 = {};
  for (int k0 = 0; k0 < K; k0 += 32) {
    v16h a  = frag_row(A, K, mrow, k0, lane);
    v16h b0 = frag_row(Wt, K, nrow, k0, lane);
    v16h b1 = frag_row(Wt, K, nrow + 16, k0, lane);
    v16h b2 = frag_row(Wt, K, nrow + 32, k0, lane);
    v16h b3 = frag_row(Wt, K, nrow + 48, k0, lane);
    acc0 = wmma16(a, b0, acc0);
    acc1 = wmma16(a, b1, acc1);
    acc2 = wmma16(a, b2, acc2);
    acc3 = wmma16(a, b3, acc3);
  }
  const int half = lane >> 4;
#pragma unroll
  for (int t = 0; t < 4; ++t) {
    v8f acc = (t == 0) ? acc0 : (t == 1) ? acc1 : (t == 2) ? acc2 : acc3;
    int n = n0 + t * 16 + (lane & 15);
    float bv = bias[n];
#pragma unroll
    for (int r = 0; r < 8; ++r) {
      float x = acc[r] + bv;
      if (act == ACT_SOFTPLUS)
        x = (x > 20.f) ? x : log1pf(expf(x));
      else if (act == ACT_SILU)
        x = x / (1.f + expf(-x));
      Out[(size_t)(m0 + r + 8 * half) * Ncols + n] = (_Float16)x;
    }
  }
}

// ---- fused edge attention: one block = 16 edges x 1 head, 4 waves ----------
// Wmu_t: 384x384 (transposed, rows contiguous in K); Wm_t: 128x384.
__global__ __launch_bounds__(128) void edge_attn(
    const int* __restrict__ dst, const int* __restrict__ src,
    const _Float16* __restrict__ q, const _Float16* __restrict__ k,
    const _Float16* __restrict__ v, const _Float16* __restrict__ eh,
    const _Float16* __restrict__ Wmu_t, const float* __restrict__ bmu,
    const _Float16* __restrict__ Wm_t, const float* __restrict__ bm,
    const float* __restrict__ lag, const float* __restrict__ lab,
    const float* __restrict__ lmg, const float* __restrict__ lmb,
    float* __restrict__ agg) {
  const int head = blockIdx.y;
  const int e0 = blockIdx.x * 16;
  const int tid = threadIdx.x;
  const int wave = tid >> 5, lane = tid & 31;

  __shared__ int s_dst[16], s_src[16];
  __shared__ float    s_kcat[16][C3];  // kcat -> sigmoid gate (in place)
  __shared__ float    s_qdo[16][C];    // q[dst]; later reused for OUT tile
  __shared__ _Float16 s_Amsg[16][C3];  // [v_dst | v_src | eh] (GEMM A)
  __shared__ _Float16 s_mu[16][C3];    // msg@Wmu+bmu; then gated in place

  if (tid < 16) {
    s_dst[tid] = dst[e0 + tid];
    s_src[tid] = src[e0 + tid];
  }
  __syncthreads();

  // gather node/edge rows into LDS (coalesced over channels)
  for (int idx = tid; idx < 16 * C; idx += 128) {
    int r = idx >> 7, c = idx & 127;
    int d = s_dst[r], s = s_src[r];
    int col = head * C + c;
    _Float16 kd = k[(size_t)d * HC + col];
    _Float16 ks = k[(size_t)s * HC + col];
    _Float16 ev = eh[(size_t)(e0 + r) * HC + col];
    s_kcat[r][c] = (float)kd;
    s_kcat[r][C + c] = (float)ks;
    s_kcat[r][2 * C + c] = (float)ev;
    s_qdo[r][c] = (float)q[(size_t)d * HC + col];
    s_Amsg[r][c]         = v[(size_t)d * HC + col];
    s_Amsg[r][C + c]     = v[(size_t)s * HC + col];
    s_Amsg[r][2 * C + c] = ev;
  }
  __syncthreads();

  // alpha = q3 * kcat / sqrt(384); LayerNorm over 384; sigmoid -> s_kcat
  {
    const float inv = 0.051031036307982884f;  // 1/sqrt(384)
    const int r = tid >> 3, sub = tid & 7;    // 8 threads per edge-row
    float lsum = 0.f, lsq = 0.f;
    for (int j0 = 0; j0 < 48; ++j0) {
      int j = sub * 48 + j0;
      float a = s_qdo[r][j & 127] * s_kcat[r][j] * inv;
      lsum += a;
      lsq += a * a;
    }
#pragma unroll
    for (int o = 4; o > 0; o >>= 1) {
      lsum += __shfl_xor(lsum, o, 8);
      lsq  += __shfl_xor(lsq, o, 8);
    }
    float mean = lsum * (1.f / 384.f);
    float var = lsq * (1.f / 384.f) - mean * mean;
    float rstd = rsqrtf(var + 1e-5f);
    for (int j0 = 0; j0 < 48; ++j0) {
      int j = sub * 48 + j0;
      float a = s_qdo[r][j & 127] * s_kcat[r][j] * inv;
      a = (a - mean) * rstd * lag[j] + lab[j];
      s_kcat[r][j] = 1.f / (1.f + expf(-a));
    }
  }
  __syncthreads();

  // MU(16x384) = Amsg @ Wmu + bmu; 4 waves split the 24 n-tiles
  for (int nt = wave; nt < C3 / 16; nt += 4) {
    v8f acc = {};
    const int nrow = nt * 16 + (lane & 15);
    for (int k0 = 0; k0 < C3; k0 += 32) {
      v16h a = frag_row(&s_Amsg[0][0], C3, lane & 15, k0, lane);
      v16h b = frag_row(Wmu_t, C3, nrow, k0, lane);
      acc = wmma16(a, b, acc);
    }
    const int half = lane >> 4, n = nt * 16 + (lane & 15);
    const float bv = bmu[n];
#pragma unroll
    for (int r = 0; r < 8; ++r) s_mu[r + 8 * half][n] = (_Float16)(acc[r] + bv);
  }
  __syncthreads();

  // gated = MU * sigmoid(LN(alpha))   (in place on s_mu)
  for (int idx = tid; idx < 16 * C3; idx += 128) {
    int r = idx / C3, j = idx - r * C3;
    s_mu[r][j] = (_Float16)((float)s_mu[r][j] * s_kcat[r][j]);
  }
  __syncthreads();

  // OUT(16x128) = gated @ Wm + bm  -> s_qdo
  for (int nt = wave; nt < C / 16; nt += 4) {
    v8f acc = {};
    const int nrow = nt * 16 + (lane & 15);
    for (int k0 = 0; k0 < C3; k0 += 32) {
      v16h a = frag_row(&s_mu[0][0], C3, lane & 15, k0, lane);
      v16h b = frag_row(Wm_t, C3, nrow, k0, lane);
      acc = wmma16(a, b, acc);
    }
    const int half = lane >> 4, n = nt * 16 + (lane & 15);
    const float bv = bm[n];
#pragma unroll
    for (int r = 0; r < 8; ++r) s_qdo[r + 8 * half][n] = acc[r] + bv;
  }
  __syncthreads();

  // LayerNorm over 128 and scatter-add into agg[dst]
  {
    const int r = tid >> 3, sub = tid & 7;
    float lsum = 0.f, lsq = 0.f;
    for (int j0 = 0; j0 < 16; ++j0) {
      float x = s_qdo[r][sub * 16 + j0];
      lsum += x;
      lsq += x * x;
    }
#pragma unroll
    for (int o = 4; o > 0; o >>= 1) {
      lsum += __shfl_xor(lsum, o, 8);
      lsq  += __shfl_xor(lsq, o, 8);
    }
    float mean = lsum * (1.f / 128.f);
    float var = lsq * (1.f / 128.f) - mean * mean;
    float rstd = rsqrtf(var + 1e-5f);
    int d = s_dst[r];
    for (int j0 = 0; j0 < 16; ++j0) {
      int j = sub * 16 + j0;
      float y = (s_qdo[r][j] - mean) * rstd * lmg[j] + lmb[j];
      atomicAdd(&agg[(size_t)d * HC + head * C + j], y);
    }
  }
}

// ---- helper kernels --------------------------------------------------------
__global__ void convert_f16(const float* __restrict__ in,
                            _Float16* __restrict__ out, int n) {
  int i = blockIdx.x * 256 + threadIdx.x;
  if (i < n) out[i] = (_Float16)in[i];
}

// Wt[n*Kp + k] = k < K ? W[k*N + n] : 0   (transpose + K-pad + f16)
__global__ void transpose_f16_pad(const float* __restrict__ W,
                                  _Float16* __restrict__ Wt, int K, int N,
                                  int Kp) {
  int i = blockIdx.x * 256 + threadIdx.x;
  if (i >= N * Kp) return;
  int n = i / Kp, kk = i - n * Kp;
  Wt[i] = (_Float16)(kk < K ? W[(size_t)kk * N + n] : 0.f);
}

// pack x (N x 92 f32) -> (N x 96 f16, zero padded)
__global__ void pack_x(const float* __restrict__ x, _Float16* __restrict__ xp) {
  int i = blockIdx.x, t = threadIdx.x;  // 96 threads
  xp[(size_t)i * AIFP + t] = (_Float16)(t < AIF ? x[(size_t)i * AIF + t] : 0.f);
}

__global__ void fill_f32(float* p, float val, int n) {
  int i = blockIdx.x * 256 + threadIdx.x;
  if (i < n) p[i] = val;
}

__global__ void rbf_kernel(const float* __restrict__ ea,
                           _Float16* __restrict__ out) {
  int e = blockIdx.x, c = threadIdx.x;
  __shared__ float ds;
  if (c == 0) {
    float a = ea[e * 3], b = ea[e * 3 + 1], cc = ea[e * 3 + 2];
    ds = sqrtf(a * a + b * b + cc * cc);
  }
  __syncthreads();
  float center = (8.0f / 127.0f) * c;
  float t = ds - center;
  out[(size_t)e * C + c] = (_Float16)expf(-(127.0f / 8.0f) * t * t);
}

__global__ void bn_stats(const _Float16* __restrict__ t,
                         float* __restrict__ sum, float* __restrict__ sumsq,
                         int M) {
  int c = blockIdx.x;
  int tid = threadIdx.x;
  float s = 0.f, s2 = 0.f;
  for (int i = tid; i < M; i += 256) {
    float x = (float)t[(size_t)i * C + c];
    s += x;
    s2 += x * x;
  }
  __shared__ float rs[256], rq[256];
  rs[tid] = s;
  rq[tid] = s2;
  __syncthreads();
  for (int st = 128; st > 0; st >>= 1) {
    if (tid < st) {
      rs[tid] += rs[tid + st];
      rq[tid] += rq[tid + st];
    }
    __syncthreads();
  }
  if (tid == 0) {
    sum[c] = rs[0];
    sumsq[c] = rq[0];
  }
}

__global__ void bn_silu(const _Float16* __restrict__ t,
                        const float* __restrict__ sum,
                        const float* __restrict__ sumsq,
                        const float* __restrict__ g,
                        const float* __restrict__ b,
                        _Float16* __restrict__ out, int M) {
  int i = blockIdx.x * 256 + threadIdx.x;
  if (i >= M * C) return;
  int c = i & 127;
  float mean = sum[c] / (float)M;
  float var = sumsq[c] / (float)M - mean * mean;
  float x = ((float)t[i] - mean) * rsqrtf(var + 1e-5f) * g[c] + b[c];
  out[i] = (_Float16)(x / (1.f + expf(-x)));
}

__global__ void pool_add(const _Float16* __restrict__ h,
                         const int* __restrict__ batch,
                         float* __restrict__ pooled, float* __restrict__ cnt) {
  int i = blockIdx.x, c = threadIdx.x;
  int g = batch[i];
  atomicAdd(&pooled[(size_t)g * C + c], (float)h[(size_t)i * C + c]);
  if (c == 0) atomicAdd(&cnt[g], 1.f);
}

__global__ void head_kernel(const float* __restrict__ pooled,
                            const float* __restrict__ cnt,
                            const float* __restrict__ fcW,
                            const float* __restrict__ fcb,
                            const float* __restrict__ outW,
                            const float* __restrict__ outb,
                            float* __restrict__ out) {
  int g = blockIdx.x, c = threadIdx.x;
  __shared__ float sp[C], sr[C];
  float cv = fmaxf(cnt[g], 1.f);
  sp[c] = pooled[(size_t)g * C + c] / cv;
  __syncthreads();
  float acc = fcb[c];
  for (int kk = 0; kk < C; ++kk) acc += sp[kk] * fcW[kk * C + c];
  float f = acc / (1.f + expf(-acc));
  sr[c] = f * outW[c];
  __syncthreads();
  for (int st = 64; st > 0; st >>= 1) {
    if (c < st) sr[c] += sr[c + st];
    __syncthreads();
  }
  if (c == 0) out[g] = sr[0] + outb[0];
}

// ---- driver ----------------------------------------------------------------
extern "C" void kernel_launch(void* const* d_in, const int* in_sizes, int n_in,
                              void* d_out, int out_size, void* d_ws,
                              size_t ws_size, hipStream_t stream) {
  (void)in_sizes; (void)n_in; (void)out_size; (void)ws_size;
  auto P = [&](int i) { return (const float*)d_in[i]; };
  const float* x = P(0);
  const float* edge_attr = P(1);
  const int* ei = (const int*)d_in[2];
  const int* batch = (const int*)d_in[3];
  const int* src = ei;            // edge_index[0]
  const int* dst = ei + N_EDGES;  // edge_index[1]

  char* base = (char*)d_ws;
  size_t off = 0;
  auto takeF = [&](size_t n) {
    float* p = (float*)(base + off);
    off += ((n * 4 + 255) & ~(size_t)255);
    return p;
  };
  auto takeH = [&](size_t n) {
    _Float16* p = (_Float16*)(base + off);
    off += ((n * 2 + 255) & ~(size_t)255);
    return p;
  };

  // f32 buffers (reductions / atomics only)
  float* agg = takeF((size_t)N_NODES * HC);
  float* bnsum = takeF(C);
  float* bnsq = takeF(C);
  float* pooled = takeF((size_t)N_GRAPH * C);
  float* cnt = takeF(N_GRAPH);

  // f16 activations
  _Float16* xp = takeH((size_t)N_NODES * AIFP);
  _Float16* h = takeH((size_t)N_NODES * C);
  _Float16* rbf = takeH((size_t)N_EDGES * C);
  _Float16* ef = takeH((size_t)N_EDGES * C);
  _Float16* qb = takeH((size_t)N_NODES * HC);
  _Float16* kb = takeH((size_t)N_NODES * HC);
  _Float16* vb = takeH((size_t)N_NODES * HC);
  _Float16* ehb = takeH((size_t)N_EDGES * HC);
  _Float16* agg16 = takeH((size_t)N_NODES * HC);
  _Float16* tmp = takeH((size_t)N_NODES * C);

  // f16 transposed weights (N x K, rows contiguous in K)
  _Float16* atomWt = takeH((size_t)C * AIFP);
  _Float16* rbfW1t = takeH((size_t)C * C);
  _Float16* rbfW2t = takeH((size_t)C * C);
  _Float16 *Wqt[NLAYERS], *Wkt[NLAYERS], *Wvt[NLAYERS], *Wet[NLAYERS];
  _Float16 *Wcatt[NLAYERS], *Wmut[NLAYERS], *Wmt[NLAYERS];
  for (int l = 0; l < NLAYERS; ++l) {
    Wqt[l] = takeH((size_t)HC * C);
    Wkt[l] = takeH((size_t)HC * C);
    Wvt[l] = takeH((size_t)HC * C);
    Wet[l] = takeH((size_t)HC * C);
    Wcatt[l] = takeH((size_t)C * HC);
    Wmut[l] = takeH((size_t)C3 * C3);
    Wmt[l] = takeH((size_t)C * C3);
  }

  auto tr = [&](const float* W, _Float16* Wt, int K, int N, int Kp) {
    int n = N * Kp;
    transpose_f16_pad<<<(n + 255) / 256, 256, 0, stream>>>(W, Wt, K, N, Kp);
  };
  tr(P(4), atomWt, AIF, C, AIFP);
  tr(P(6), rbfW1t, C, C, C);
  tr(P(8), rbfW2t, C, C, C);
  for (int l = 0; l < NLAYERS; ++l) {
    int b0 = 14 + l * 20;
    tr(P(b0 + 0), Wqt[l], C, HC, C);
    tr(P(b0 + 2), Wkt[l], C, HC, C);
    tr(P(b0 + 4), Wvt[l], C, HC, C);
    tr(P(b0 + 6), Wet[l], C, HC, C);
    tr(P(b0 + 8), Wcatt[l], HC, C, HC);
    tr(P(b0 + 10), Wmut[l], C3, C3, C3);
    tr(P(b0 + 12), Wmt[l], C3, C, C3);
  }

  auto gemm = [&](const _Float16* A, const _Float16* Wt, const float* bias,
                  _Float16* Out, int M, int K, int Ncol, int act) {
    gemm16<<<dim3(M / 16, Ncol / 64), 32, 0, stream>>>(A, Wt, bias, Out, M, K,
                                                       Ncol, act);
  };

  // atom embedding: h = x @ atom_W + atom_b   (K padded 92 -> 96)
  pack_x<<<N_NODES, AIFP, 0, stream>>>(x, xp);
  gemm(xp, atomWt, P(5), h, N_NODES, AIFP, C, ACT_NONE);

  // RBF expansion + 2-layer MLP -> ef  (ehb reused as scratch)
  rbf_kernel<<<N_EDGES, C, 0, stream>>>(edge_attr, rbf);
  gemm(rbf, rbfW1t, P(7), ehb, N_EDGES, C, C, ACT_SOFTPLUS);
  gemm(ehb, rbfW2t, P(9), ef, N_EDGES, C, C, ACT_NONE);

  for (int l = 0; l < NLAYERS; ++l) {
    int b0 = 14 + l * 20;
    gemm(h, Wqt[l], P(b0 + 1), qb, N_NODES, C, HC, ACT_NONE);
    gemm(h, Wkt[l], P(b0 + 3), kb, N_NODES, C, HC, ACT_NONE);
    gemm(h, Wvt[l], P(b0 + 5), vb, N_NODES, C, HC, ACT_NONE);
    gemm(ef, Wet[l], P(b0 + 7), ehb, N_EDGES, C, HC, ACT_NONE);

    fill_f32<<<((N_NODES * HC) + 255) / 256, 256, 0, stream>>>(agg, 0.f,
                                                               N_NODES * HC);

    edge_attn<<<dim3(N_EDGES / 16, H), 128, 0, stream>>>(
        dst, src, qb, kb, vb, ehb, Wmut[l], P(b0 + 11), Wmt[l], P(b0 + 13),
        P(b0 + 14), P(b0 + 15), P(b0 + 16), P(b0 + 17), agg);

    convert_f16<<<((N_NODES * HC) + 255) / 256, 256, 0, stream>>>(
        agg, agg16, N_NODES * HC);
    gemm(agg16, Wcatt[l], P(b0 + 9), tmp, N_NODES, HC, C, ACT_NONE);
    bn_stats<<<C, 256, 0, stream>>>(tmp, bnsum, bnsq, N_NODES);
    bn_silu<<<((N_NODES * C) + 255) / 256, 256, 0, stream>>>(
        tmp, bnsum, bnsq, P(b0 + 18), P(b0 + 19), h, N_NODES);
  }

  fill_f32<<<((N_GRAPH * C) + 255) / 256, 256, 0, stream>>>(pooled, 0.f,
                                                            N_GRAPH * C);
  fill_f32<<<1, 256, 0, stream>>>(cnt, 0.f, N_GRAPH);
  pool_add<<<N_NODES, C, 0, stream>>>(h, batch, pooled, cnt);
  head_kernel<<<N_GRAPH, C, 0, stream>>>(pooled, cnt, P(10), P(11), P(12),
                                         P(13), (float*)d_out);
}
